// ScaledDotProdAtt_91199335563985
// MI455X (gfx1250) — matrix operand order repaired and checked by reading, not compiled, not run on hardware
//
#include <hip/hip_runtime.h>
#include <hip/hip_bf16.h>

// ---- problem constants (from reference) ----
#define BB   4
#define HH   16
#define SS   2048
#define DKK  128
#define NEGV (-1.0e9f)
// score scale folded into log2 domain: (1/sqrt(128)) * log2(e)
#define SC2  (0.08838834764831845f * 1.4426950408889634f)

#if __has_builtin(__builtin_amdgcn_exp2f)
#define EXP2(x) __builtin_amdgcn_exp2f(x)
#else
#define EXP2(x) exp2f(x)
#endif

typedef __attribute__((ext_vector_type(16))) __bf16 v16bf;
typedef __attribute__((ext_vector_type(8)))  __bf16 v8bf;
typedef __attribute__((ext_vector_type(4)))  __bf16 v4bf;
typedef __attribute__((ext_vector_type(8)))  float  v8f;

// ------------------------------------------------------------------
// Prologue 1: K (f32) -> K (bf16), same [bh][s][d] layout.
// ------------------------------------------------------------------
__global__ void cvt_bf16_kernel(const float* __restrict__ in,
                                __bf16* __restrict__ out, long n4) {
  long i = (long)blockIdx.x * blockDim.x + threadIdx.x;
  long stride = (long)gridDim.x * blockDim.x;
  for (; i < n4; i += stride) {
    float4 f = ((const float4*)in)[i];
    v4bf o;
    o[0] = (__bf16)f.x; o[1] = (__bf16)f.y;
    o[2] = (__bf16)f.z; o[3] = (__bf16)f.w;
    ((v4bf*)out)[i] = o;
  }
}

// ------------------------------------------------------------------
// Prologue 2: V (f32, [bh][s][d]) -> Vt (bf16, [bh][d][s])
// LDS-tiled 32x32 transpose; coalesced read & write.
// ------------------------------------------------------------------
__global__ void transpose_v_kernel(const float* __restrict__ v,
                                   __bf16* __restrict__ vt) {
  __shared__ float tile[32][33];
  int stile = blockIdx.x;          // S/32 tiles
  int dtile = blockIdx.y;          // DK/32 tiles
  int bh    = blockIdx.z;
  int tx = threadIdx.x;            // 0..31
  int ty = threadIdx.y;            // 0..7
  const float* src = v + ((size_t)bh * SS) * DKK;
#pragma unroll
  for (int i = 0; i < 4; ++i) {
    int s = stile * 32 + ty * 4 + i;
    tile[ty * 4 + i][tx] = src[(size_t)s * DKK + dtile * 32 + tx];
  }
  __syncthreads();
  __bf16* dst = vt + ((size_t)bh * DKK) * SS;
#pragma unroll
  for (int i = 0; i < 4; ++i) {
    int d = dtile * 32 + ty * 4 + i;
    dst[(size_t)d * SS + stile * 32 + tx] = (__bf16)tile[tx][ty * 4 + i];
  }
}

// ------------------------------------------------------------------
// 16-lane row reductions (rows of C layout live in lanes [0,15] / [16,31])
// ------------------------------------------------------------------
__device__ __forceinline__ float rowmax16(float x) {
#pragma unroll
  for (int m = 1; m < 16; m <<= 1) x = fmaxf(x, __shfl_xor(x, m, 32));
  return x;
}
__device__ __forceinline__ float rowsum16(float x) {
#pragma unroll
  for (int m = 1; m < 16; m <<= 1) x += __shfl_xor(x, m, 32);
  return x;
}

// ------------------------------------------------------------------
// Main flash-attention kernel. One wave per 16-query tile, 4 waves/block,
// 64 keys per iteration (32 WMMAs per softmax update).
// ------------------------------------------------------------------
#define WPB 4
__global__ void __launch_bounds__(32 * WPB)
fa_wmma_kernel(const float* __restrict__ q,
               const int*  __restrict__ sen_len,
               const __bf16* __restrict__ kbf,   // [bh][s][d] bf16
               const __bf16* __restrict__ vt,    // [bh][d][s] bf16
               float* __restrict__ out) {
  __shared__ __bf16 ldsP[WPB * 16 * 64];        // per-wave P staging (2KB each)

  const int lane = threadIdx.x & 31;
  const int wave = threadIdx.x >> 5;
  const int ln   = lane & 15;      // N index within tile
  const int half = lane >> 4;      // lane half (selects K/M sub-block)

  const int tiles_per_bh = SS / 16;                 // 128
  const int blocks_per_bh = tiles_per_bh / WPB;     // 32
  const int bh = blockIdx.x / blocks_per_bh;
  const int qtile = (blockIdx.x % blocks_per_bh) * WPB + wave;
  const int qbase = qtile * 16;
  const int b = bh / HH;
  const int slen = sen_len[b];

  __bf16* Pb = &ldsP[wave * (16 * 64)];

  // ---- load Q tile (16 x 128) as 4 bf16 A-fragments (16x32 each) ----
  // A layout: lane row m = ln; element f -> k = half*16+f
  const float* qrow = q + ((size_t)bh * SS + qbase + ln) * DKK;
  v16bf qa[4];
#pragma unroll
  for (int t = 0; t < 4; ++t) {
    int k0 = 32 * t + half * 16;
    float4 a0 = *(const float4*)(qrow + k0);
    float4 a1 = *(const float4*)(qrow + k0 + 4);
    float4 b0 = *(const float4*)(qrow + k0 + 8);
    float4 b1 = *(const float4*)(qrow + k0 + 12);
    qa[t][0] = (__bf16)a0.x; qa[t][1] = (__bf16)a0.y;
    qa[t][2] = (__bf16)a0.z; qa[t][3] = (__bf16)a0.w;
    qa[t][4] = (__bf16)a1.x; qa[t][5] = (__bf16)a1.y;
    qa[t][6] = (__bf16)a1.z; qa[t][7] = (__bf16)a1.w;
    qa[t][8]  = (__bf16)b0.x; qa[t][9]  = (__bf16)b0.y;
    qa[t][10] = (__bf16)b0.z; qa[t][11] = (__bf16)b0.w;
    qa[t][12] = (__bf16)b1.x; qa[t][13] = (__bf16)b1.y;
    qa[t][14] = (__bf16)b1.z; qa[t][15] = (__bf16)b1.w;
  }

  // ---- accumulators / running softmax state (log2 domain) ----
  v8f o[8];
#pragma unroll
  for (int d = 0; d < 8; ++d) o[d] = (v8f)0.0f;
  float mrun[8], lrun[8];
  int rlim[8];  // per-row key limit: key <= rlim passes both masks
#pragma unroll
  for (int e = 0; e < 8; ++e) {
    mrun[e] = -1.0e30f;
    lrun[e] = 0.0f;
    int r = qbase + half * 8 + e;
    rlim[e] = (r < slen) ? ((r < slen - 1) ? r : (slen - 1)) : -1;
  }

  const int nsteps = (qbase + 16 + 63) >> 6;   // causal: keys 0..qbase+15
  const __bf16* kbase_ptr = kbf + ((size_t)bh * SS) * DKK;
  const __bf16* vbase_ptr = vt + ((size_t)bh * DKK) * SS;

  for (int step = 0; step < nsteps; ++step) {
    const int kb = step * 64;

    // prefetch next step's K/V lines into near caches
    if (step + 1 < nsteps) {
      __builtin_prefetch(kbase_ptr + (size_t)(kb + 64 + lane * 2) * DKK, 0, 0);
      __builtin_prefetch(vbase_ptr + (size_t)(lane * 4) * SS + kb + 64, 0, 0);
    }

    // ---- QK^T : four 16x16 C tiles over 64 keys ----
    v8f s[4];
#pragma unroll
    for (int c = 0; c < 4; ++c) s[c] = (v8f)0.0f;
#pragma unroll
    for (int c = 0; c < 4; ++c) {
      const __bf16* krow = kbase_ptr + (size_t)(kb + 16 * c + ln) * DKK + half * 16;
#pragma unroll
      for (int t = 0; t < 4; ++t) {
        v16bf kfrag = *(const v16bf*)(krow + 32 * t);
        s[c] = __builtin_amdgcn_wmma_f32_16x16x32_bf16(false, qa[t], false, kfrag,
                                                       (short)0, s[c], false, false);
      }
    }

    // ---- scale (log2 domain) + fused causal/padding mask ----
#pragma unroll
    for (int c = 0; c < 4; ++c) {
      const int col = kb + 16 * c + ln;
#pragma unroll
      for (int e = 0; e < 8; ++e) {
        float a = s[c][e] * SC2;
        s[c][e] = (col <= rlim[e]) ? a : NEGV;
      }
    }

    // ---- online softmax update (one reduction per 64 keys) ----
    float alpha[8];
#pragma unroll
    for (int e = 0; e < 8; ++e) {
      float mx = fmaxf(fmaxf(s[0][e], s[1][e]), fmaxf(s[2][e], s[3][e]));
      float nm = rowmax16(fmaxf(mrun[e], mx));
      alpha[e] = EXP2(mrun[e] - nm);
      float p0 = EXP2(s[0][e] - nm);
      float p1 = EXP2(s[1][e] - nm);
      float p2 = EXP2(s[2][e] - nm);
      float p3 = EXP2(s[3][e] - nm);
      s[0][e] = p0; s[1][e] = p1; s[2][e] = p2; s[3][e] = p3;
      lrun[e] = alpha[e] * lrun[e] + rowsum16((p0 + p1) + (p2 + p3));
      mrun[e] = nm;
    }
#pragma unroll
    for (int d = 0; d < 8; ++d)
#pragma unroll
      for (int e = 0; e < 8; ++e) o[d][e] *= alpha[e];

    // ---- P: C layout -> A layout via wave-private LDS (16 x 64 bf16) ----
#pragma unroll
    for (int e = 0; e < 8; ++e) {
      int rl = half * 8 + e;
#pragma unroll
      for (int c = 0; c < 4; ++c)
        Pb[rl * 64 + 16 * c + ln] = (__bf16)s[c][e];
    }
    v16bf pa[2];
#pragma unroll
    for (int j = 0; j < 2; ++j) {
      v8bf plo = *(const v8bf*)&Pb[ln * 64 + 32 * j + half * 16];
      v8bf phi = *(const v8bf*)&Pb[ln * 64 + 32 * j + half * 16 + 8];
#pragma unroll
      for (int i = 0; i < 8; ++i) { pa[j][i] = plo[i]; pa[j][8 + i] = phi[i]; }
    }

    // ---- P @ V : 8 dim-tiles x 2 key-halves; B = Vt[d][kb..kb+63] ----
#pragma unroll
    for (int d = 0; d < 8; ++d) {
      const __bf16* vrow = vbase_ptr + (size_t)(d * 16 + ln) * SS + kb + half * 16;
#pragma unroll
      for (int j = 0; j < 2; ++j) {
        v16bf vb = *(const v16bf*)(vrow + 32 * j);
        o[d] = __builtin_amdgcn_wmma_f32_16x16x32_bf16(false, pa[j], false, vb,
                                                       (short)0, o[d], false, false);
      }
    }
  }

  // ---- epilogue: normalize + store (f32 out, C layout mapping) ----
  float inv[8];
#pragma unroll
  for (int e = 0; e < 8; ++e) inv[e] = 1.0f / lrun[e];
  float* orow = out + ((size_t)bh * SS + qbase) * DKK;
#pragma unroll
  for (int d = 0; d < 8; ++d) {
#pragma unroll
    for (int e = 0; e < 8; ++e) {
      int r = half * 8 + e;
      orow[(size_t)r * DKK + d * 16 + ln] = o[d][e] * inv[e];
    }
  }
}

// ------------------------------------------------------------------
// host launch
// ------------------------------------------------------------------
extern "C" void kernel_launch(void* const* d_in, const int* in_sizes, int n_in,
                              void* d_out, int out_size, void* d_ws, size_t ws_size,
                              hipStream_t stream) {
  const float* q = (const float*)d_in[0];
  const float* k = (const float*)d_in[1];
  const float* v = (const float*)d_in[2];
  const int* sen_len = (const int*)d_in[3];
  float* out = (float*)d_out;

  const size_t nelem = (size_t)BB * HH * SS * DKK;   // 16.78M
  __bf16* kbf = (__bf16*)d_ws;
  __bf16* vtr = kbf + nelem;

  // 1) K -> bf16
  cvt_bf16_kernel<<<2048, 256, 0, stream>>>(k, kbf, (long)(nelem / 4));
  // 2) V -> bf16 transposed [bh][d][s]
  transpose_v_kernel<<<dim3(SS / 32, DKK / 32, BB * HH), dim3(32, 8), 0, stream>>>(v, vtr);
  // 3) flash attention
  const int blocks = BB * HH * (SS / 16) / WPB;      // 2048
  fa_wmma_kernel<<<blocks, 32 * WPB, 0, stream>>>(q, sen_len, kbf, vtr, out);
}
